// ScannedRNN_2551210574113
// MI455X (gfx1250) — matrix operand order repaired and checked by reading, not compiled, not run on hardware
//
#include <hip/hip_runtime.h>

// GRU scan for MI455X (gfx1250): persistent cooperative kernel, one grid-wide
// barrier per timestep, bf16 WMMA 16x16x32 for both GEMMs.
// - Weights: converted to bf16 once, staged in LDS in B-operand layout (48 KB).
// - h carry: f32 ping-pong buffer (exact z*h term) + packed bf16 A-operand
//   layout ping-pong buffer (feeds recurrent WMMA with zero per-step VALU).
// - x: converted on the fly with single-instruction v_perm_b32 bf16 packing.
//
// T=1024, B=256, H=256. Grid = 32 blocks x 256 threads (8 waves).
// Block (blockIdx & 15) -> 16-column tile; (blockIdx >> 4) -> 128-row half of
// the batch; wave w -> 16-row tile. Each wave owns one 16x16 h_new tile and
// accumulates 6 gate tiles (ir,iz,in,hr,hz,hn), K=256 in 8 chunks of 32.

#define T_STEPS 1024
#define BATCH   256
#define HID     256
#define H3      768
#define NBLK    32
#define NTHR    256

typedef __attribute__((ext_vector_type(16))) __bf16 v16bf;
typedef __attribute__((ext_vector_type(8)))  float  v8f;

union ABOp { unsigned u[8]; v16bf v; };

// One-instruction bf16 pair pack (truncation): v_perm_b32 selects the high
// 16 bits of each f32. dst = { hi16(b_hi), hi16(a_lo) }.
__device__ __forceinline__ unsigned pack_trunc(float lo, float hi) {
  return __builtin_amdgcn_perm(__float_as_uint(hi), __float_as_uint(lo), 0x07060302u);
}

__device__ __forceinline__ float sigmoid_f(float x) {
  return 1.0f / (1.0f + __expf(-x));
}
__device__ __forceinline__ float tanh_f(float x) {
  return 1.0f - 2.0f / (1.0f + __expf(2.0f * x));  // saturates correctly
}

// B-operand fetch: pre-swizzled in LDS so a lane's 8 dwords are contiguous.
__device__ __forceinline__ v16bf load_b(const unsigned* lds, int g, int kc, int lane) {
  ABOp b;
  const uint4* p = (const uint4*)(lds + (((g * 8 + kc) * 32 + lane) * 8));
  uint4 lo = p[0], hi = p[1];
  b.u[0] = lo.x; b.u[1] = lo.y; b.u[2] = lo.z; b.u[3] = lo.w;
  b.u[4] = hi.x; b.u[5] = hi.y; b.u[6] = hi.z; b.u[7] = hi.w;
  return b.v;
}

// Grid barrier: monotonic counter, release on arrive, acquire spin.
__device__ __forceinline__ void grid_barrier(unsigned* cnt, unsigned target) {
  __syncthreads();
  if (threadIdx.x == 0) {
    __threadfence();
    __hip_atomic_fetch_add(cnt, 1u, __ATOMIC_ACQ_REL, __HIP_MEMORY_SCOPE_AGENT);
    while (__hip_atomic_load(cnt, __ATOMIC_ACQUIRE, __HIP_MEMORY_SCOPE_AGENT) < target) {
      __builtin_amdgcn_s_sleep(1);
    }
  }
  __syncthreads();
}

// Packed bf16 A-layout: per row, 128 dwords; dword (kc,hf,j) holds K pair
// K = kc*32 + (j<4 ? hf*8+2j : 16+hf*8+2(j-4)). A lane (m-row, half) then
// reads dwords [kc*16 + half*8 .. +7] as two b128 loads.
__global__ void __launch_bounds__(NTHR) gru_init_kernel(const float* __restrict__ h0,
                                                        float* __restrict__ hA,
                                                        unsigned* __restrict__ hbfA,
                                                        unsigned* __restrict__ cnt) {
  int i = blockIdx.x * blockDim.x + threadIdx.x;
  if (i == 0) *cnt = 0u;
  if (i < BATCH * HID) hA[i] = h0[i];
  if (i < BATCH * 128) {
    int row = i >> 7, d = i & 127;
    int kc = d >> 4, r = d & 15, hf = r >> 3, j = r & 7;
    int K = kc * 32 + ((j < 4) ? (hf * 8 + 2 * j) : (16 + hf * 8 + 2 * (j - 4)));
    hbfA[i] = pack_trunc(h0[row * HID + K], h0[row * HID + K + 1]);
  }
}

__global__ void __launch_bounds__(NTHR) gru_scan_kernel(
    const float* __restrict__ x,            // [T,B,H]
    const unsigned char* __restrict__ rst,  // [T,B] bool (1 byte)
    const float* __restrict__ Wi,           // [H,3H]
    const float* __restrict__ bi,           // [3H]
    const float* __restrict__ Wh,           // [H,3H]
    const float* __restrict__ bhn,          // [H]
    float* __restrict__ out,                // [T,B,H]
    float* __restrict__ hA, float* __restrict__ hB,          // f32 carry ping/pong
    unsigned* __restrict__ hbfA, unsigned* __restrict__ hbfB, // bf16 A-layout ping/pong
    unsigned* __restrict__ cnt)
{
  // [gate 0..5][k-chunk 0..7][lane 0..31][dword 0..7] : 48 KB
  __shared__ unsigned lds_w[6 * 8 * 32 * 8];

  const int tid  = threadIdx.x;
  const int lane = tid & 31;
  const int wave = tid >> 5;
  const int n    = lane & 15;   // output column within tile (B / C / D layout)
  const int half = lane >> 4;   // K-half for A/B, M-half for C/D
  const int c0   = (blockIdx.x & 15) * 16;
  const int r0   = (blockIdx.x >> 4) * 128 + wave * 16;
  const int mrow = r0 + (lane & 15);      // A-operand row this lane feeds

  // ---- Stage all 6 weight gate-slices as bf16 into B-operand layout (once).
  for (int idx = tid; idx < 6 * 8 * 32 * 8; idx += NTHR) {
    int g   = idx >> 11;
    int rem = idx & 2047;
    int kc  = rem >> 8;
    int l2  = (rem >> 3) & 31;
    int j   = rem & 7;
    int nn2 = l2 & 15, hh2 = l2 >> 4;
    int k   = kc * 32 + ((j < 4) ? (hh2 * 8 + 2 * j) : (16 + hh2 * 8 + 2 * (j - 4)));
    int col = (g % 3) * HID + c0 + nn2;
    const float* W = (g < 3) ? Wi : Wh;
    lds_w[idx] = pack_trunc(W[(size_t)k * H3 + col], W[(size_t)(k + 1) * H3 + col]);
  }
  __syncthreads();

  const float bir   = bi[c0 + n];
  const float biz   = bi[HID + c0 + n];
  const float bin   = bi[2 * HID + c0 + n];
  const float bhn_l = bhn[c0 + n];

  // Step-invariant byte offset of this lane's output column inside a packed
  // bf16 A-layout row (see gru_init_kernel layout comment).
  const int c   = c0 + n;
  const int kcx = c >> 5, qx = c & 31, px = qx >> 4, q2 = qx & 15;
  const int hfx = q2 >> 3, ox = q2 & 7;
  const int jjx = (ox >> 1) + (px ? 4 : 0);
  const int bofs = (kcx * 16 + hfx * 8 + jjx) * 4 + (ox & 1) * 2;

  for (int t = 0; t < T_STEPS; ++t) {
    const float*    hb_cur  = (t & 1) ? hB : hA;
    float*          hb_nxt  = (t & 1) ? hA : hB;
    const unsigned* hbf_cur = (t & 1) ? hbfB : hbfA;
    unsigned*       hbf_nxt = (t & 1) ? hbfA : hbfB;

    const float*    xrow    = x + ((size_t)t * BATCH + mrow) * HID;
    const unsigned* hbf_row = hbf_cur + (size_t)mrow * 128;
    const unsigned  rmask   = rst[t * BATCH + mrow] ? 0u : 0xFFFFFFFFu;  // A-side reset

    if (t + 1 < T_STEPS) __builtin_prefetch(xrow + (size_t)BATCH * HID, 0, 1);

    v8f acc_ir = {0.f,0.f,0.f,0.f,0.f,0.f,0.f,0.f};
    v8f acc_iz = {0.f,0.f,0.f,0.f,0.f,0.f,0.f,0.f};
    v8f acc_in = {0.f,0.f,0.f,0.f,0.f,0.f,0.f,0.f};
    v8f acc_hr = {0.f,0.f,0.f,0.f,0.f,0.f,0.f,0.f};
    v8f acc_hz = {0.f,0.f,0.f,0.f,0.f,0.f,0.f,0.f};
    v8f acc_hn = {0.f,0.f,0.f,0.f,0.f,0.f,0.f,0.f};

#pragma unroll
    for (int kc = 0; kc < 8; ++kc) {
      // x A-operand: 4x b128 f32 loads + 8 v_perm packs.
      const int kA0 = kc * 32 + half * 8;
      const int kA1 = kA0 + 16;
      float4 x0 = *(const float4*)(xrow + kA0);
      float4 x1 = *(const float4*)(xrow + kA0 + 4);
      float4 x2 = *(const float4*)(xrow + kA1);
      float4 x3 = *(const float4*)(xrow + kA1 + 4);
      ABOp ax;
      ax.u[0] = pack_trunc(x0.x, x0.y); ax.u[1] = pack_trunc(x0.z, x0.w);
      ax.u[2] = pack_trunc(x1.x, x1.y); ax.u[3] = pack_trunc(x1.z, x1.w);
      ax.u[4] = pack_trunc(x2.x, x2.y); ax.u[5] = pack_trunc(x2.z, x2.w);
      ax.u[6] = pack_trunc(x3.x, x3.y); ax.u[7] = pack_trunc(x3.z, x3.w);

      // h A-operand: already bf16-packed in A layout; 2x b128 loads + mask.
      const uint4* hp4 = (const uint4*)(hbf_row + kc * 16 + half * 8);
      uint4 hlo = hp4[0], hhi = hp4[1];
      ABOp ah;
      ah.u[0] = hlo.x & rmask; ah.u[1] = hlo.y & rmask;
      ah.u[2] = hlo.z & rmask; ah.u[3] = hlo.w & rmask;
      ah.u[4] = hhi.x & rmask; ah.u[5] = hhi.y & rmask;
      ah.u[6] = hhi.z & rmask; ah.u[7] = hhi.w & rmask;

      acc_ir = __builtin_amdgcn_wmma_f32_16x16x32_bf16(false, ax.v, false, load_b(lds_w, 0, kc, lane), (short)0, acc_ir, false, false);
      acc_iz = __builtin_amdgcn_wmma_f32_16x16x32_bf16(false, ax.v, false, load_b(lds_w, 1, kc, lane), (short)0, acc_iz, false, false);
      acc_in = __builtin_amdgcn_wmma_f32_16x16x32_bf16(false, ax.v, false, load_b(lds_w, 2, kc, lane), (short)0, acc_in, false, false);
      acc_hr = __builtin_amdgcn_wmma_f32_16x16x32_bf16(false, ah.v, false, load_b(lds_w, 3, kc, lane), (short)0, acc_hr, false, false);
      acc_hz = __builtin_amdgcn_wmma_f32_16x16x32_bf16(false, ah.v, false, load_b(lds_w, 4, kc, lane), (short)0, acc_hz, false, false);
      acc_hn = __builtin_amdgcn_wmma_f32_16x16x32_bf16(false, ah.v, false, load_b(lds_w, 5, kc, lane), (short)0, acc_hn, false, false);
    }

    // ---- Elementwise gates. C/D layout: lane (half,n), VGPR v -> M = v + 8*half.
#pragma unroll
    for (int v = 0; v < 8; ++v) {
      int row  = r0 + v + 8 * half;
      int hidx = row * HID + c0 + n;
      float hp = rst[t * BATCH + row] ? 0.0f : hb_cur[hidx];  // masked f32 carry
      float rr  = sigmoid_f(acc_ir[v] + bir + acc_hr[v]);
      float zz  = sigmoid_f(acc_iz[v] + biz + acc_hz[v]);
      float nnv = tanh_f(acc_in[v] + bin + rr * (acc_hn[v] + bhn_l));
      float hnew = (1.0f - zz) * nnv + zz * hp;
      hb_nxt[hidx] = hnew;
      out[(size_t)t * (BATCH * HID) + hidx] = hnew;
      // bf16 copy into next step's A-operand layout (single b16 store).
      *(unsigned short*)((char*)(hbf_nxt + (size_t)row * 128) + bofs) =
          (unsigned short)(__float_as_uint(hnew) >> 16);
    }

    grid_barrier(cnt, (unsigned)(t + 1) * NBLK);
  }
}

extern "C" void kernel_launch(void* const* d_in, const int* in_sizes, int n_in,
                              void* d_out, int out_size, void* d_ws, size_t ws_size,
                              hipStream_t stream) {
  (void)in_sizes; (void)n_in; (void)out_size; (void)ws_size;
  const float*         x    = (const float*)d_in[0];
  const unsigned char* rst  = (const unsigned char*)d_in[1];  // numpy bool = 1 byte
  const float*         h0   = (const float*)d_in[2];
  const float*         Wi   = (const float*)d_in[3];
  const float*         bi   = (const float*)d_in[4];
  const float*         Wh   = (const float*)d_in[5];
  const float*         bhn  = (const float*)d_in[6];
  float*               out  = (float*)d_out;

  char* ws = (char*)d_ws;
  float*    hA   = (float*)ws;                                   // 256 KB
  float*    hB   = (float*)(ws + 262144);                        // 256 KB
  unsigned* hbfA = (unsigned*)(ws + 524288);                     // 128 KB
  unsigned* hbfB = (unsigned*)(ws + 524288 + 131072);            // 128 KB
  unsigned* cnt  = (unsigned*)(ws + 524288 + 262144);

  gru_init_kernel<<<(BATCH * HID + NTHR - 1) / NTHR, NTHR, 0, stream>>>(h0, hA, hbfA, cnt);
  gru_scan_kernel<<<NBLK, NTHR, 0, stream>>>(x, rst, Wi, bi, Wh, bhn, out,
                                             hA, hB, hbfA, hbfB, cnt);
}
